// RNN_73907797229590
// MI455X (gfx1250) — matrix operand order repaired and checked by reading, not compiled, not run on hardware
//
#include <hip/hip_runtime.h>
#include <hip/hip_bf16.h>

// ---------------------------------------------------------------------------
// Seq2seq LSTM (B=256, S=120, T=24, D=216, H=1024) on gfx1250.
// Recurrent + input GEMMs via v_wmma_f32_16x16x32_bf16 (bf16 in, fp32 acc).
// One kernel launch per timestep; gate GEMM + LSTM cell fused.
// A-panel ([x,h] 16 x 1248) staged once in LDS; K loop is barrier-free.
// ---------------------------------------------------------------------------

#define BATCH 256
#define SEQ   120
#define TDEC  24
#define DIN   216
#define DPAD  224          // DIN padded to multiple of 32
#define HID   1024
#define KCAT  (DPAD + HID) // 1248
#define NC    (KCAT / 32)  // 39 K-chunks for gate GEMM
#define NC2   (HID / 32)   // 32 K-chunks for output projection
#define DPITCH 1256        // LDS A-panel row stride (ushorts), bank-friendly

typedef __attribute__((ext_vector_type(16))) __bf16 v16bf;
typedef __attribute__((ext_vector_type(8)))  float  v8f;

union Frag {
    v16bf v;
    unsigned int u[8];
};

__device__ __forceinline__ unsigned short f2bf(float x) {
    unsigned int u = __builtin_bit_cast(unsigned int, x);
    unsigned int r = u + 0x7FFFu + ((u >> 16) & 1u);   // round-to-nearest-even
    return (unsigned short)(r >> 16);
}

__device__ __forceinline__ float sigmf(float x) {
    return 1.0f / (1.0f + __expf(-x));
}

// ---------------------------------------------------------------------------
// Pack [4H x (DPAD+H)] combined gate weights (W_ih | W_hh) into bf16 in the
// WMMA B-fragment lane order this file uses (A loads match): block (ct, kc) holds 512 bf16; lane L owns 16
// contiguous bf16 = column n=L%16, K ascending (K base 16 for lanes 16..31).
// ---------------------------------------------------------------------------
__global__ __launch_bounds__(256) void pack_gate_weights(
    const float* __restrict__ W_ih, const float* __restrict__ W_hh,
    unsigned short* __restrict__ Wpack)
{
    long p = (long)blockIdx.x * 256 + threadIdx.x;       // total 256*39*512
    if (p >= (long)256 * NC * 512) return;
    int e    = (int)(p & 15);
    int lane = (int)((p >> 4) & 31);
    int kc   = (int)((p >> 9) % NC);
    int jt   = (int)(p / ((long)NC * 512));
    int k_local = ((lane >> 4) << 4) | e;
    int n  = lane & 15;
    int r  = jt * 16 + n;                                // weight row (= out col)
    int k  = kc * 32 + k_local;                          // 0..1247
    float v;
    if (k < DPAD) v = (k < DIN) ? W_ih[(long)r * DIN + k] : 0.0f;
    else          v = W_hh[(long)r * HID + (k - DPAD)];
    Wpack[p] = f2bf(v);
}

// Pack W_out [DIN x H] -> [DPAD x H] bf16, same fragment-major order.
__global__ __launch_bounds__(256) void pack_out_weights(
    const float* __restrict__ W_out, unsigned short* __restrict__ WoutP)
{
    long p = (long)blockIdx.x * 256 + threadIdx.x;       // total 14*32*512
    if (p >= (long)14 * NC2 * 512) return;
    int e    = (int)(p & 15);
    int lane = (int)((p >> 4) & 31);
    int kc   = (int)((p >> 9) % NC2);
    int jt   = (int)(p / ((long)NC2 * 512));
    int k_local = ((lane >> 4) << 4) | e;
    int n  = lane & 15;
    int r  = jt * 16 + n;                                // 0..223
    int k  = kc * 32 + k_local;                          // 0..1023
    float v = (r < DIN) ? W_out[(long)r * HID + k] : 0.0f;
    WoutP[p] = f2bf(v);
}

// ---------------------------------------------------------------------------
// One LSTM timestep, fully fused: gates = [x,h] @ Wcat^T + b ; cell update.
// Grid: (64, 16) workgroups of 128 threads (4 waves). Wave w = gate w.
// Workgroup (jt, mt) produces h/c tile rows mt*16.., cols jt*16.. .
// ---------------------------------------------------------------------------
__global__ __launch_bounds__(128) void lstm_step(
    const float*          __restrict__ xsrc,   // [B,S,D] fp32 (or null)
    const unsigned short* __restrict__ xfb,    // [B,DPAD] bf16 feedback (or null)
    int s,                                     // src frame index when xsrc!=null
    const unsigned short* __restrict__ hin,    // [B,H] bf16
    unsigned short*       __restrict__ hout,   // [B,H] bf16
    float*                __restrict__ cbuf,   // [B,H] fp32
    const unsigned short* __restrict__ Wpack,  // packed gate weights
    const float* __restrict__ b_ih, const float* __restrict__ b_hh)
{
    __shared__ unsigned short Abuf[16][DPITCH]; // full [x,h] A-panel (bf16)
    __shared__ float Gbuf[4][16][16];           // per-gate 16x16 fp32 tiles

    const int tid  = threadIdx.x;
    const int wave = tid >> 5;                 // gate id 0..3 (i,f,g,o)
    const int lane = tid & 31;
    const int jt   = blockIdx.x;               // hidden col tile 0..63
    const int mt   = blockIdx.y;               // batch row tile 0..15
    const int ct   = wave * 64 + jt;           // global gate-col tile 0..255
    const int n    = lane & 15;
    const int hi   = lane >> 4;
    const int kA   = hi * 16;

    // Cooperative staging indices: 8 threads per A row.
    const int am   = tid >> 3;                 // row 0..15
    const int aq   = tid & 7;
    const int arow = mt * 16 + am;

    // ---- stage x part (7 chunks of 32 bf16; DIN=216 is quad-aligned) ----
    if (xsrc) {
        const float* sp = xsrc + ((long)arow * SEQ + s) * DIN;
        #pragma unroll
        for (int kc = 0; kc < 7; ++kc) {
            const int kb = kc * 32 + aq * 4;
            union { unsigned short s4[4]; uint2 u; } t;
            if (kb < DIN) {                    // all 4 elements valid
                t.s4[0] = f2bf(sp[kb + 0]);
                t.s4[1] = f2bf(sp[kb + 1]);
                t.s4[2] = f2bf(sp[kb + 2]);
                t.s4[3] = f2bf(sp[kb + 3]);
            } else {
                t.u.x = 0u; t.u.y = 0u;       // padding 216..223
            }
            *(uint2*)&Abuf[am][kb] = t.u;
        }
    } else {
        const unsigned short* sp = xfb + (long)arow * DPAD;
        #pragma unroll
        for (int kc = 0; kc < 7; ++kc) {
            const int kb = kc * 32 + aq * 4;
            *(uint2*)&Abuf[am][kb] = *(const uint2*)(sp + kb);
        }
    }
    // ---- stage h part: 2048 B/row, 8 threads/row, 16 x b128 each ----
    {
        const uint4* hp = (const uint4*)(hin + (long)arow * HID);
        #pragma unroll
        for (int j = 0; j < 16; ++j) {
            uint4 q = hp[aq + j * 8];
            *(uint4*)&Abuf[am][DPAD + (aq + j * 8) * 8] = q;
        }
    }
    __syncthreads();

    // ---- barrier-free WMMA K loop: 39 chunks of K=32 ----
    const unsigned short* arow_p = &Abuf[n][0];          // A row M = n
    const unsigned short* bbase  = Wpack + (long)ct * NC * 512 + lane * 16;

    v8f acc = {};
    for (int kc = 0; kc < NC; ++kc) {
        Frag a;
        {
            uint4 q0 = *(const uint4*)(arow_p + kc * 32 + kA);       // K=kA..+7
            uint4 q1 = *(const uint4*)(arow_p + kc * 32 + kA + 8);   // K=kA+8..+15
            a.u[0] = q0.x; a.u[1] = q0.y; a.u[2] = q0.z; a.u[3] = q0.w;
            a.u[4] = q1.x; a.u[5] = q1.y; a.u[6] = q1.z; a.u[7] = q1.w;
        }
        Frag b;
        {
            const uint4* bp = (const uint4*)(bbase + (long)kc * 512);
            uint4 q0 = bp[0], q1 = bp[1];
            b.u[0] = q0.x; b.u[1] = q0.y; b.u[2] = q0.z; b.u[3] = q0.w;
            b.u[4] = q1.x; b.u[5] = q1.y; b.u[6] = q1.z; b.u[7] = q1.w;
        }
        if (kc + 1 < NC)
            __builtin_prefetch((const void*)(bbase + (long)(kc + 1) * 512), 0, 3);

        acc = __builtin_amdgcn_wmma_f32_16x16x32_bf16(
            false, a.v, false, b.v, (short)0, acc, false, false);
    }

    // ---- bias + stash per-gate tile ----
    const int colg = (wave << 10) + (jt << 4) + n;
    const float bias = b_ih[colg] + b_hh[colg];
    #pragma unroll
    for (int r = 0; r < 8; ++r)
        Gbuf[wave][r + hi * 8][n] = acc[r] + bias;
    __syncthreads();

    // ---- fused LSTM cell: 128 threads cover 256 elements ----
    #pragma unroll
    for (int t2 = 0; t2 < 2; ++t2) {
        const int idx = tid + t2 * 128;
        const int m = idx >> 4, nn = idx & 15;
        const float ig = Gbuf[0][m][nn];
        const float fg = Gbuf[1][m][nn];
        const float gg = Gbuf[2][m][nn];
        const float og = Gbuf[3][m][nn];
        const long off = ((long)mt * 16 + m) * HID + (long)jt * 16 + nn;
        const float cold = cbuf[off];
        const float cnew = sigmf(fg) * cold + sigmf(ig) * tanhf(gg);
        cbuf[off] = cnew;
        hout[off] = f2bf(sigmf(og) * tanhf(cnew));
    }
}

// ---------------------------------------------------------------------------
// Decoder output projection: out = h @ W_out^T + b_out.
// Grid (14, 16), 1 wave/WG, 16x16 output tile, K=1024 in 32 WMMA chunks.
// A fragment loaded straight from global h with 2x b128 (16B-aligned).
// ---------------------------------------------------------------------------
__global__ __launch_bounds__(32) void proj_step(
    const unsigned short* __restrict__ h,      // [B,H] bf16 (new h)
    const unsigned short* __restrict__ WoutP,
    const float* __restrict__ b_out,
    float* __restrict__ out,                   // [B,T,D] fp32
    unsigned short* __restrict__ fb,           // [B,DPAD] bf16 feedback
    int tstep)
{
    const int lane = threadIdx.x;
    const int jt = blockIdx.x;                 // 0..13 output col tile
    const int mt = blockIdx.y;                 // 0..15 batch row tile
    const int n  = lane & 15;
    const int hi = lane >> 4;
    const int kA = hi * 16;

    const unsigned short* hrow  = h + ((long)mt * 16 + n) * HID;      // M = n
    const unsigned short* bbase = WoutP + (long)jt * NC2 * 512 + lane * 16;

    v8f acc = {};
    for (int kc = 0; kc < NC2; ++kc) {
        Frag a;
        {
            uint4 q0 = *(const uint4*)(hrow + kc * 32 + kA);
            uint4 q1 = *(const uint4*)(hrow + kc * 32 + kA + 8);
            a.u[0] = q0.x; a.u[1] = q0.y; a.u[2] = q0.z; a.u[3] = q0.w;
            a.u[4] = q1.x; a.u[5] = q1.y; a.u[6] = q1.z; a.u[7] = q1.w;
        }
        Frag b;
        {
            const uint4* bp = (const uint4*)(bbase + (long)kc * 512);
            uint4 q0 = bp[0], q1 = bp[1];
            b.u[0] = q0.x; b.u[1] = q0.y; b.u[2] = q0.z; b.u[3] = q0.w;
            b.u[4] = q1.x; b.u[5] = q1.y; b.u[6] = q1.z; b.u[7] = q1.w;
        }
        if (kc + 1 < NC2)
            __builtin_prefetch((const void*)(bbase + (long)(kc + 1) * 512), 0, 3);

        acc = __builtin_amdgcn_wmma_f32_16x16x32_bf16(
            false, a.v, false, b.v, (short)0, acc, false, false);
    }

    const int col = jt * 16 + n;
    if (col < DIN) {                           // divergence only after WMMAs
        const float bias = b_out[col];
        #pragma unroll
        for (int r = 0; r < 8; ++r) {
            const long row = (long)mt * 16 + r + hi * 8;
            const float v = acc[r] + bias;
            out[(row * TDEC + tstep) * DIN + col] = v;
            fb[row * DPAD + col] = f2bf(v);    // padding cols stay zeroed
        }
    }
}

// ---------------------------------------------------------------------------
// Workspace layout (bytes):
//   hbuf0  @ 0         524288   bf16 [B,H]
//   hbuf1  @ 524288    524288
//   cbuf   @ 1048576   1048576  f32  [B,H]
//   fb0    @ 2097152   114688   bf16 [B,DPAD]
//   fb1    @ 2211840   114688
//   Wpack  @ 2326528   10223616 bf16 fragment-major gate weights
//   WoutP  @ 12550144  458752   bf16 fragment-major output weights
//   total ~12.4 MB
// ---------------------------------------------------------------------------
extern "C" void kernel_launch(void* const* d_in, const int* in_sizes, int n_in,
                              void* d_out, int out_size, void* d_ws, size_t ws_size,
                              hipStream_t stream) {
    (void)in_sizes; (void)n_in; (void)out_size; (void)ws_size;
    const float* src   = (const float*)d_in[0];
    // d_in[1] = tgt: unused (eval-mode autoregressive decoder)
    const float* W_ih  = (const float*)d_in[2];
    const float* W_hh  = (const float*)d_in[3];
    const float* b_ih  = (const float*)d_in[4];
    const float* b_hh  = (const float*)d_in[5];
    const float* W_out = (const float*)d_in[6];
    const float* b_out = (const float*)d_in[7];
    float* out = (float*)d_out;

    char* ws = (char*)d_ws;
    unsigned short* hbuf[2] = {(unsigned short*)(ws + 0),
                               (unsigned short*)(ws + 524288)};
    float*          cbuf    =  (float*)         (ws + 1048576);
    unsigned short* fb[2]   = {(unsigned short*)(ws + 2097152),
                               (unsigned short*)(ws + 2211840)};
    unsigned short* Wpack   =  (unsigned short*)(ws + 2326528);
    unsigned short* WoutP   =  (unsigned short*)(ws + 12550144);

    // zero h0, c0 and feedback padding (graph-capturable memset node)
    hipMemsetAsync(ws, 0, 2326528, stream);

    pack_gate_weights<<<19968, 256, 0, stream>>>(W_ih, W_hh, Wpack);
    pack_out_weights <<<  896, 256, 0, stream>>>(W_out, WoutP);

    const dim3 sg(64, 16);
    // Encoder: teacher-forced over src frames
    for (int s = 0; s < SEQ; ++s)
        lstm_step<<<sg, 128, 0, stream>>>(src, nullptr, s,
                                          hbuf[s & 1], hbuf[(s + 1) & 1],
                                          cbuf, Wpack, b_ih, b_hh);
    // Decoder: autoregressive rollout, first input = src[:, -1, :]
    for (int t = 0; t < TDEC; ++t) {
        const float*          xs = (t == 0) ? src : nullptr;
        const unsigned short* xf = (t == 0) ? nullptr : fb[t & 1];
        lstm_step<<<sg, 128, 0, stream>>>(xs, xf, SEQ - 1,
                                          hbuf[t & 1], hbuf[(t + 1) & 1],
                                          cbuf, Wpack, b_ih, b_hh);
        proj_step<<<dim3(14, 16), 32, 0, stream>>>(hbuf[(t + 1) & 1], WoutP,
                                                   b_out, out, fb[(t + 1) & 1], t);
    }
}
